// PointTransformer_18562848653514
// MI455X (gfx1250) — matrix operand order repaired and checked by reading, not compiled, not run on hardware
//
#include <hip/hip_runtime.h>
#include <hip/hip_bf16.h>

#define N_NODES 20000
#define N_EDGES 640000

typedef __attribute__((ext_vector_type(16))) __bf16         v16bf;
typedef __attribute__((ext_vector_type(8)))  float          v8f;
typedef __attribute__((ext_vector_type(4)))  unsigned short u16x4;
typedef __attribute__((ext_vector_type(8)))  unsigned short u16x8;
typedef __attribute__((ext_vector_type(16))) unsigned short u16x16;

// ---------- helpers ----------
__device__ __forceinline__ unsigned short f32_to_bf16u(float f) {
  unsigned int u = __builtin_bit_cast(unsigned int, f);
  u += 0x7FFFu + ((u >> 16) & 1u);  // round-to-nearest-even
  return (unsigned short)(u >> 16);
}
__device__ __forceinline__ v8f wmma_bf16(v16bf a, v16bf b, v8f c) {
  return __builtin_amdgcn_wmma_f32_16x16x32_bf16(false, a, false, b, (short)0, c,
                                                 false, false);
}
__device__ __forceinline__ v16bf combine16(u16x8 lo, u16x8 hi) {
  u16x16 w = __builtin_shufflevector(lo, hi, 0, 1, 2, 3, 4, 5, 6, 7, 8, 9, 10, 11, 12,
                                     13, 14, 15);
  return __builtin_bit_cast(v16bf, w);
}

// LDS byte offset of a pointer into a __shared__ object (for async-to-LDS dst)
__device__ __forceinline__ unsigned lds_offset(const void* p) {
  return (unsigned)(unsigned long long)(__attribute__((address_space(3))) const void*)p;
}

// A fragment 16x32 bf16 from LDS (row-major, stride lda).
// CDNA5 16-bit A layout is K-contiguous per lane: e<8 -> K=half*8+e,
// e>=8 -> K=16+half*8+(e-8)  => two aligned 16-byte ds loads.
__device__ __forceinline__ v16bf load_a_frag_lds(const unsigned short* A, int row0,
                                                 int k0, int lda) {
  const int l = threadIdx.x & 31, half = l >> 4, m = l & 15;
  const unsigned short* p = A + (row0 + m) * lda + k0 + half * 8;
  u16x8 lo = *(const u16x8*)p;
  u16x8 hi = *(const u16x8*)(p + 16);
  return combine16(lo, hi);
}

// B fragment from pre-packed bf16 weights (fragment order): lane's 16 elements are
// 32 contiguous bytes => two coalesced global 16-byte loads, no conversion.
__device__ __forceinline__ v16bf load_b_frag_p(const unsigned short* __restrict__ Wp,
                                               int tile) {
  const int l = threadIdx.x & 31;
  const unsigned short* p = Wp + (((tile << 5) + l) << 4);
  u16x8 lo = *(const u16x8*)p;
  u16x8 hi = *(const u16x8*)(p + 8);
  return combine16(lo, hi);
}

// order-preserving float<->uint encoding for atomic f32 max via atomicMax(u32)
__device__ __forceinline__ unsigned enc_f32(float f) {
  unsigned u = __builtin_bit_cast(unsigned, f);
  return (u & 0x80000000u) ? ~u : (u | 0x80000000u);
}
__device__ __forceinline__ float dec_f32(unsigned u) {
  unsigned b = (u & 0x80000000u) ? (u & 0x7FFFFFFFu) : ~u;
  return __builtin_bit_cast(float, b);
}

// ---------- off-path packing kernels ----------

// W[K,N] f32 -> bf16 fragment-order: out[((kc*NTN+nt)*32 + lane)*16 + e],
// element = W[kc*32 + (e>>3)*16 + half*8 + (e&7)][nt*16 + n], zero-padded.
__global__ void pack_w(const float* __restrict__ W, int K, int N, int KCN, int NTN,
                       unsigned short* __restrict__ out) {
  int idx = blockIdx.x * 256 + threadIdx.x;
  if (idx >= KCN * NTN * 512) return;
  int e = idx & 15;
  int lane = (idx >> 4) & 31;
  int tile = idx >> 9;
  int nt = tile % NTN;
  int kc = tile / NTN;
  int k = kc * 32 + (e >> 3) * 16 + (lane >> 4) * 8 + (e & 7);
  int col = nt * 16 + (lane & 15);
  float v = (k < K && col < N) ? W[k * N + col] : 0.0f;
  out[idx] = f32_to_bf16u(v);
}

// f32 node features -> bf16 [N,128] zero-padded (optional ReLU), 4 elems/thread
__global__ void pack_nodes(const float* __restrict__ in, int in_stride, int din,
                           int do_relu, unsigned short* __restrict__ out) {
  int g = blockIdx.x * 256 + threadIdx.x;  // group of 4 columns
  if (g >= N_NODES * 32) return;
  int n = g >> 5, c = (g & 31) << 2;
  u16x4 o = {0, 0, 0, 0};
  if (c < din) {
    const float* p = in + n * in_stride + c;
    float a = p[0], b = p[1], d = p[2], e = p[3];
    if (do_relu) {
      a = fmaxf(a, 0.f); b = fmaxf(b, 0.f); d = fmaxf(d, 0.f); e = fmaxf(e, 0.f);
    }
    o = u16x4{f32_to_bf16u(a), f32_to_bf16u(b), f32_to_bf16u(d), f32_to_bf16u(e)};
  }
  *(u16x4*)(out + n * 128 + c) = o;
}

// ---------- node GEMM ----------
// Y[N,128] f32 = Xb[N,128] bf16 @ Wp (packed). 8 waves; wave = 16-row tile.
// N_NODES % 16 == 0 => tile validity is wave-uniform: no per-element store guards.
template <int DIN, int DOUT>
__global__ __launch_bounds__(256) void node_gemm(const unsigned short* __restrict__ X,
                                                 const unsigned short* __restrict__ Wp,
                                                 float* __restrict__ Y) {
  constexpr int KCN = (DIN + 31) / 32;
  constexpr int NTN = (DOUT + 15) / 16;
  __shared__ unsigned short ldsX[128 * 128];  // 32 KB
  const int r0 = blockIdx.x * 128;
  // stage the 128x128 bf16 slab with gfx1250 async global->LDS DMA (16B per lane).
  // OOB rows clamp to the last valid row; their tiles are skipped below.
  for (int i = threadIdx.x * 8; i < 128 * 128; i += 256 * 8) {
    int r = i >> 7, c = i & 127;
    int gr = r0 + r;
    if (gr >= N_NODES) gr = N_NODES - 1;
    unsigned long long ga = (unsigned long long)(X + (size_t)gr * 128 + c);
    unsigned ldst = lds_offset(&ldsX[i]);
    asm volatile("global_load_async_to_lds_b128 %0, %1, off"
                 :
                 : "v"(ldst), "v"(ga)
                 : "memory");
  }
  asm volatile("s_wait_asynccnt 0" ::: "memory");
  __syncthreads();
  const int wave = threadIdx.x >> 5, lane = threadIdx.x & 31;
  const int half = lane >> 4, n = lane & 15;
  const int row0 = wave * 16;
  if (r0 + row0 < N_NODES) {  // wave-uniform tile bound
    v16bf afrag[KCN];
#pragma unroll
    for (int kc = 0; kc < KCN; ++kc)
      afrag[kc] = load_a_frag_lds(ldsX, row0, kc * 32, 128);
#pragma unroll
    for (int nt = 0; nt < NTN; ++nt) {
      v8f acc;
#pragma unroll
      for (int r = 0; r < 8; ++r) acc[r] = 0.0f;
#pragma unroll
      for (int kc = 0; kc < KCN; ++kc)
        acc = wmma_bf16(afrag[kc], load_b_frag_p(Wp, kc * NTN + nt), acc);
#pragma unroll
      for (int r = 0; r < 8; ++r)
        Y[(r0 + row0 + half * 8 + r) * 128 + nt * 16 + n] = acc[r];
    }
  }
}

// ---------- edge pos MLP: 2 -> 64 (VALU) -> 64 -> DOUT ----------
template <int DOUT>
__global__ __launch_bounds__(256) void edge_pos_mlp(
    const float* __restrict__ pos, const int* __restrict__ src,
    const int* __restrict__ dst, const float* __restrict__ W1,
    const float* __restrict__ B1, const float* __restrict__ B2,
    const float* __restrict__ B3, const unsigned short* __restrict__ W2p,
    const unsigned short* __restrict__ W3p, float* __restrict__ delta) {
  constexpr int NTN3 = (DOUT + 15) / 16;
  __shared__ float pd[8][32];
  __shared__ unsigned short h1[8][16 * 64];
  __shared__ unsigned short h2[8][16 * 64];
  const int wave = threadIdx.x >> 5, lane = threadIdx.x & 31;
  const int half = lane >> 4, n = lane & 15;
  // layer-1 weights: this lane only ever touches cols lane and lane+32
  const float w0c0 = W1[lane], w0c1 = W1[lane + 32];
  const float w1c0 = W1[64 + lane], w1c1 = W1[96 + lane];
  const float b1c0 = B1[lane], b1c1 = B1[lane + 32];
  float bv2[4], bv3[NTN3];
#pragma unroll
  for (int nt = 0; nt < 4; ++nt) bv2[nt] = B2[nt * 16 + n];
#pragma unroll
  for (int nt = 0; nt < NTN3; ++nt) {
    int col = nt * 16 + n;
    bv3[nt] = (col < DOUT) ? B3[col] : 0.0f;
  }
  const int ntiles = N_EDGES / 16;
  for (int t = blockIdx.x * 8 + wave; t < ntiles; t += gridDim.x * 8) {
    const int e0 = t * 16;
    {  // pos[dst]-pos[src]: lane -> (row = n, component = half)
      int e = e0 + n;
      pd[wave][n * 2 + half] = pos[dst[e] * 2 + half] - pos[src[e] * 2 + half];
    }
    // layer1: 2 -> 64, pure FMA
#pragma unroll
    for (int r = 0; r < 16; ++r) {
      float p0 = pd[wave][r * 2], p1 = pd[wave][r * 2 + 1];
      h1[wave][r * 64 + lane] = f32_to_bf16u(fmaxf(p0 * w0c0 + p1 * w1c0 + b1c0, 0.f));
      h1[wave][r * 64 + lane + 32] =
          f32_to_bf16u(fmaxf(p0 * w0c1 + p1 * w1c1 + b1c1, 0.f));
    }
    // layer2: 64 -> 64 WMMA
#pragma unroll
    for (int nt = 0; nt < 4; ++nt) {
      v8f acc;
#pragma unroll
      for (int r = 0; r < 8; ++r) acc[r] = bv2[nt];
#pragma unroll
      for (int kc = 0; kc < 2; ++kc)
        acc = wmma_bf16(load_a_frag_lds(h1[wave], 0, kc * 32, 64),
                        load_b_frag_p(W2p, kc * 4 + nt), acc);
#pragma unroll
      for (int r = 0; r < 8; ++r)
        h2[wave][(half * 8 + r) * 64 + nt * 16 + n] = f32_to_bf16u(fmaxf(acc[r], 0.f));
    }
    // layer3: 64 -> DOUT, store delta (stride 128)
#pragma unroll
    for (int nt = 0; nt < NTN3; ++nt) {
      v8f acc;
#pragma unroll
      for (int r = 0; r < 8; ++r) acc[r] = bv3[nt];
#pragma unroll
      for (int kc = 0; kc < 2; ++kc)
        acc = wmma_bf16(load_a_frag_lds(h2[wave], 0, kc * 32, 64),
                        load_b_frag_p(W3p, kc * NTN3 + nt), acc);
#pragma unroll
      for (int r = 0; r < 8; ++r)
        delta[(size_t)(e0 + half * 8 + r) * 128 + nt * 16 + n] = acc[r];
    }
  }
}

// ---------- edge attn MLP: u = a_dst[dst]-a_src[src]+delta, D -> 64 -> 64 -> D ----------
template <int D>
__global__ __launch_bounds__(256) void edge_attn_mlp(
    const float* __restrict__ Asrc, const float* __restrict__ Adst,
    const float* __restrict__ delta, const int* __restrict__ src,
    const int* __restrict__ dst, const float* __restrict__ B1,
    const float* __restrict__ B2, const float* __restrict__ B3,
    const unsigned short* __restrict__ W1p, const unsigned short* __restrict__ W2p,
    const unsigned short* __restrict__ W3p, float* __restrict__ alpha) {
  constexpr int KP = (D + 31) & ~31;
  constexpr int KCN1 = KP / 32;
  constexpr int NTN3 = (D + 15) / 16;
  __shared__ unsigned short u[8][16 * KP];
  __shared__ unsigned short h1[8][16 * 64];
  __shared__ unsigned short h2[8][16 * 64];
  __shared__ int sid[8][16];
  __shared__ int did[8][16];
  const int wave = threadIdx.x >> 5, lane = threadIdx.x & 31;
  const int half = lane >> 4, n = lane & 15;
  float bv1[4], bv2[4], bv3[NTN3];
#pragma unroll
  for (int nt = 0; nt < 4; ++nt) bv1[nt] = B1[nt * 16 + n];
#pragma unroll
  for (int nt = 0; nt < 4; ++nt) bv2[nt] = B2[nt * 16 + n];
#pragma unroll
  for (int nt = 0; nt < NTN3; ++nt) {
    int col = nt * 16 + n;
    bv3[nt] = (col < D) ? B3[col] : 0.0f;
  }
  const int ntiles = N_EDGES / 16;
  for (int t = blockIdx.x * 8 + wave; t < ntiles; t += gridDim.x * 8) {
    const int e0 = t * 16;
    if (half == 0) sid[wave][n] = src[e0 + n];
    else           did[wave][n] = dst[e0 + n];
    // input tile [16][KP] bf16, 4 cols per lane-iter, float4 gathers (L2-resident)
#pragma unroll
    for (int i = lane; i < (16 * KP) >> 2; i += 32) {
      int row = (i << 2) / KP;
      int col = (i << 2) & (KP - 1);
      u16x4 o = {0, 0, 0, 0};
      if (col < D) {
        int e = e0 + row;
        const float* pd2 = Adst + did[wave][row] * 128 + col;
        const float* ps2 = Asrc + sid[wave][row] * 128 + col;
        const float* pl2 = delta + (size_t)e * 128 + col;
        o = u16x4{f32_to_bf16u(pd2[0] - ps2[0] + pl2[0]),
                  f32_to_bf16u(pd2[1] - ps2[1] + pl2[1]),
                  f32_to_bf16u(pd2[2] - ps2[2] + pl2[2]),
                  f32_to_bf16u(pd2[3] - ps2[3] + pl2[3])};
      }
      *(u16x4*)(&u[wave][row * KP + col]) = o;
    }
    // layer1: D -> 64
#pragma unroll
    for (int nt = 0; nt < 4; ++nt) {
      v8f acc;
#pragma unroll
      for (int r = 0; r < 8; ++r) acc[r] = bv1[nt];
#pragma unroll
      for (int kc = 0; kc < KCN1; ++kc)
        acc = wmma_bf16(load_a_frag_lds(u[wave], 0, kc * 32, KP),
                        load_b_frag_p(W1p, kc * 4 + nt), acc);
#pragma unroll
      for (int r = 0; r < 8; ++r)
        h1[wave][(half * 8 + r) * 64 + nt * 16 + n] = f32_to_bf16u(fmaxf(acc[r], 0.f));
    }
    // layer2: 64 -> 64
#pragma unroll
    for (int nt = 0; nt < 4; ++nt) {
      v8f acc;
#pragma unroll
      for (int r = 0; r < 8; ++r) acc[r] = bv2[nt];
#pragma unroll
      for (int kc = 0; kc < 2; ++kc)
        acc = wmma_bf16(load_a_frag_lds(h1[wave], 0, kc * 32, 64),
                        load_b_frag_p(W2p, kc * 4 + nt), acc);
#pragma unroll
      for (int r = 0; r < 8; ++r)
        h2[wave][(half * 8 + r) * 64 + nt * 16 + n] = f32_to_bf16u(fmaxf(acc[r], 0.f));
    }
    // layer3: 64 -> D, store alpha (stride 128)
#pragma unroll
    for (int nt = 0; nt < NTN3; ++nt) {
      v8f acc;
#pragma unroll
      for (int r = 0; r < 8; ++r) acc[r] = bv3[nt];
#pragma unroll
      for (int kc = 0; kc < 2; ++kc)
        acc = wmma_bf16(load_a_frag_lds(h2[wave], 0, kc * 32, 64),
                        load_b_frag_p(W3p, kc * NTN3 + nt), acc);
#pragma unroll
      for (int r = 0; r < 8; ++r)
        alpha[(size_t)(e0 + half * 8 + r) * 128 + nt * 16 + n] = acc[r];
    }
  }
}

// ---------- segment softmax + aggregation ----------
__global__ void init_seg(unsigned* __restrict__ mb, float* __restrict__ s,
                         float* __restrict__ outn) {
  int tid = blockIdx.x * 256 + threadIdx.x;
  if (tid >= N_NODES * 32) return;  // 4 elems/thread
  uint4* m4 = (uint4*)mb;
  float4* s4 = (float4*)s;
  float4* o4 = (float4*)outn;
  m4[tid] = uint4{0x007FFFFFu, 0x007FFFFFu, 0x007FFFFFu, 0x007FFFFFu};  // enc(-inf)
  s4[tid] = float4{0.f, 0.f, 0.f, 0.f};
  o4[tid] = float4{0.f, 0.f, 0.f, 0.f};
}

template <int D>
__global__ void seg_max(const float* __restrict__ alpha, const int* __restrict__ dst,
                        unsigned* __restrict__ mb) {
  int tid = blockIdx.x * 256 + threadIdx.x;
  if (tid >= N_EDGES * D) return;
  int e = tid / D, c = tid & (D - 1);
  atomicMax(&mb[dst[e] * 128 + c], enc_f32(alpha[(size_t)e * 128 + c]));
}

template <int D>
__global__ void seg_expsum(float* __restrict__ alpha, const int* __restrict__ dst,
                           const unsigned* __restrict__ mb, float* __restrict__ s) {
  int tid = blockIdx.x * 256 + threadIdx.x;
  if (tid >= N_EDGES * D) return;
  int e = tid / D, c = tid & (D - 1);
  int di = dst[e] * 128 + c;
  float ev = __expf(alpha[(size_t)e * 128 + c] - dec_f32(mb[di]));
  alpha[(size_t)e * 128 + c] = ev;
  atomicAdd(&s[di], ev);
}

template <int D>
__global__ void aggregate(const float* __restrict__ ev, const float* __restrict__ s,
                          const float* __restrict__ h, const float* __restrict__ delta,
                          const int* __restrict__ src, const int* __restrict__ dst,
                          float* __restrict__ outn) {
  int tid = blockIdx.x * 256 + threadIdx.x;
  if (tid >= N_EDGES * D) return;
  int e = tid / D, c = tid & (D - 1);
  int di = dst[e] * 128 + c;
  float attn = ev[(size_t)e * 128 + c] / (s[di] + 1e-16f);
  atomicAdd(&outn[di], attn * (h[src[e] * 128 + c] + delta[(size_t)e * 128 + c]));
}

__global__ void final_copy(const float* __restrict__ outn, float* __restrict__ y) {
  int tid = blockIdx.x * 256 + threadIdx.x;
  if (tid >= N_NODES * 2) return;
  int n = tid >> 1, g = (tid & 1) << 2;
  *(float4*)(y + n * 8 + g) = *(const float4*)(outn + n * 128 + g);
}

// ---------- host ----------
struct ConvP {
  const float *aW1, *aB1, *aW2, *aB2, *aW3, *aB3;
  const float *lin, *lin_dst, *lin_src;
  const float *pW1, *pB1, *pW2, *pB2, *pW3, *pB3;
};
struct Bufs {
  unsigned short *xb, *linP, *linSrcP, *linDstP, *aW1P, *aW2P, *aW3P, *pW2P, *pW3P;
  float *h, *a_src, *a_dst, *nout, *sbuf, *delta, *alpha;
  unsigned* mb;
  const float* pos;
  const int *src, *dst;
};

template <int DIN, int DOUT>
static void run_layer(const ConvP& cv, const Bufs& B, const float* xin, int xin_stride,
                      int relu, hipStream_t stream) {
  constexpr int KCN = (DIN + 31) / 32;
  constexpr int NTN = (DOUT + 15) / 16;
  constexpr int KCN1 = ((DOUT + 31) & ~31) / 32;  // attn layer1 K-chunks
  auto pk = [&](const float* W, int K, int N, int kcn, int ntn, unsigned short* out) {
    pack_w<<<(kcn * ntn * 512 + 255) / 256, 256, 0, stream>>>(W, K, N, kcn, ntn, out);
  };
  pack_nodes<<<(N_NODES * 32 + 255) / 256, 256, 0, stream>>>(xin, xin_stride, DIN, relu,
                                                             B.xb);
  pk(cv.lin, DIN, DOUT, KCN, NTN, B.linP);
  pk(cv.lin_src, DIN, DOUT, KCN, NTN, B.linSrcP);
  pk(cv.lin_dst, DIN, DOUT, KCN, NTN, B.linDstP);
  pk(cv.aW1, DOUT, 64, KCN1, 4, B.aW1P);
  pk(cv.aW2, 64, 64, 2, 4, B.aW2P);
  pk(cv.aW3, 64, DOUT, 2, NTN, B.aW3P);
  pk(cv.pW2, 64, 64, 2, 4, B.pW2P);
  pk(cv.pW3, 64, DOUT, 2, NTN, B.pW3P);

  const int gemmGrid = (N_NODES + 127) / 128;
  node_gemm<DIN, DOUT><<<gemmGrid, 256, 0, stream>>>(B.xb, B.linP, B.h);
  node_gemm<DIN, DOUT><<<gemmGrid, 256, 0, stream>>>(B.xb, B.linSrcP, B.a_src);
  node_gemm<DIN, DOUT><<<gemmGrid, 256, 0, stream>>>(B.xb, B.linDstP, B.a_dst);

  edge_pos_mlp<DOUT><<<640, 256, 0, stream>>>(B.pos, B.src, B.dst, cv.pW1, cv.pB1,
                                              cv.pB2, cv.pB3, B.pW2P, B.pW3P, B.delta);
  edge_attn_mlp<DOUT><<<512, 256, 0, stream>>>(B.a_src, B.a_dst, B.delta, B.src, B.dst,
                                               cv.aB1, cv.aB2, cv.aB3, B.aW1P, B.aW2P,
                                               B.aW3P, B.alpha);

  init_seg<<<(N_NODES * 32 + 255) / 256, 256, 0, stream>>>(B.mb, B.sbuf, B.nout);
  const int eg = ((size_t)N_EDGES * DOUT + 255) / 256;
  seg_max<DOUT><<<eg, 256, 0, stream>>>(B.alpha, B.dst, B.mb);
  seg_expsum<DOUT><<<eg, 256, 0, stream>>>(B.alpha, B.dst, B.mb, B.sbuf);
  aggregate<DOUT><<<eg, 256, 0, stream>>>(B.alpha, B.sbuf, B.h, B.delta, B.src, B.dst,
                                          B.nout);
}

extern "C" void kernel_launch(void* const* d_in, const int* in_sizes, int n_in,
                              void* d_out, int out_size, void* d_ws, size_t ws_size,
                              hipStream_t stream) {
  (void)in_sizes; (void)n_in; (void)out_size; (void)ws_size;
  const float* x = (const float*)d_in[0];
  const float* pos = (const float*)d_in[1];
  const int* ei = (const int*)d_in[2];

  auto grab = [&](int b) {
    ConvP c;
    c.aW1 = (const float*)d_in[b + 0];  c.aB1 = (const float*)d_in[b + 1];
    c.aW2 = (const float*)d_in[b + 2];  c.aB2 = (const float*)d_in[b + 3];
    c.aW3 = (const float*)d_in[b + 4];  c.aB3 = (const float*)d_in[b + 5];
    c.lin = (const float*)d_in[b + 6];
    c.lin_dst = (const float*)d_in[b + 7];
    c.lin_src = (const float*)d_in[b + 8];
    c.pW1 = (const float*)d_in[b + 9];  c.pB1 = (const float*)d_in[b + 10];
    c.pW2 = (const float*)d_in[b + 11]; c.pB2 = (const float*)d_in[b + 12];
    c.pW3 = (const float*)d_in[b + 13]; c.pB3 = (const float*)d_in[b + 14];
    return c;
  };
  // JAX pytree leaf order: params dict keys sorted -> hidden[0]@3, hidden[1]@18,
  // input@33, output@48 (15 leaves per conv; bias=False lin layers are not leaves)
  ConvP L0 = grab(33), L1 = grab(3), L2 = grab(18), L3 = grab(48);

  char* w = (char*)d_ws;
  auto carve = [&](size_t bytes) {
    void* p = (void*)w;
    w += (bytes + 255) & ~(size_t)255;
    return p;
  };
  Bufs B;
  B.pos = pos;
  B.src = ei;
  B.dst = ei + N_EDGES;
  B.xb = (unsigned short*)carve((size_t)N_NODES * 128 * 2);
  B.h = (float*)carve((size_t)N_NODES * 128 * 4);
  B.a_src = (float*)carve((size_t)N_NODES * 128 * 4);
  B.a_dst = (float*)carve((size_t)N_NODES * 128 * 4);
  B.nout = (float*)carve((size_t)N_NODES * 128 * 4);
  B.mb = (unsigned*)carve((size_t)N_NODES * 128 * 4);
  B.sbuf = (float*)carve((size_t)N_NODES * 128 * 4);
  B.linP = (unsigned short*)carve(4 * 8 * 512 * 2);
  B.linSrcP = (unsigned short*)carve(4 * 8 * 512 * 2);
  B.linDstP = (unsigned short*)carve(4 * 8 * 512 * 2);
  B.aW1P = (unsigned short*)carve(4 * 4 * 512 * 2);
  B.aW2P = (unsigned short*)carve(2 * 4 * 512 * 2);
  B.aW3P = (unsigned short*)carve(2 * 8 * 512 * 2);
  B.pW2P = (unsigned short*)carve(2 * 4 * 512 * 2);
  B.pW3P = (unsigned short*)carve(2 * 8 * 512 * 2);
  B.delta = (float*)carve((size_t)N_EDGES * 128 * 4);
  B.alpha = (float*)carve((size_t)N_EDGES * 128 * 4);

  run_layer<32, 128>(L0, B, x, 32, 0, stream);
  run_layer<128, 128>(L1, B, B.nout, 128, 1, stream);
  run_layer<128, 128>(L2, B, B.nout, 128, 1, stream);
  run_layer<128, 8>(L3, B, B.nout, 128, 1, stream);

  final_copy<<<(N_NODES * 2 + 255) / 256, 256, 0, stream>>>(B.nout, (float*)d_out);
}
